// Transducer_29884382445666
// MI455X (gfx1250) — matrix operand order repaired and checked by reading, not compiled, not run on hardware
//
#include <hip/hip_runtime.h>
#include <math.h>

#define BD   4
#define TD   256
#define UD   64
#define ED   640
#define HD   512
#define VD   4096
#define BEAM 3

typedef _Float16 v16h __attribute__((ext_vector_type(16)));
typedef _Float16 v8h  __attribute__((ext_vector_type(8)));
typedef _Float16 v4h  __attribute__((ext_vector_type(4)));
typedef float    v8f  __attribute__((ext_vector_type(8)));

union HF16 { v16h v; v8h h[2]; };

// ---------------------------------------------------------------------------
// Kernel 0: convert W2 (V x H, f32 row-major) -> f16.  4MB result, L2-resident.
// ---------------------------------------------------------------------------
__global__ __launch_bounds__(256) void cvt_w2_kernel(const float* __restrict__ W2,
                                                     _Float16* __restrict__ w2h) {
    int gid  = blockIdx.x * 256 + threadIdx.x;
    int base = gid * 4;
    if (base >= VD * HD) return;
    float4 v = *(const float4*)(W2 + base);
    v4h h = { (_Float16)v.x, (_Float16)v.y, (_Float16)v.z, (_Float16)v.w };
    *(v4h*)(w2h + base) = h;
}

// ---------------------------------------------------------------------------
// Kernel 1: projections.  encp[bt,h] = enc[bt,:]·W1[h,0:640] + b1[h]
//                         decp[bu,h] = dec[bu,:]·W1[h,640:1280]
// ---------------------------------------------------------------------------
__global__ __launch_bounds__(256) void proj_kernel(const float* __restrict__ enc,
                                                   const float* __restrict__ dec,
                                                   const float* __restrict__ W1,
                                                   const float* __restrict__ b1,
                                                   float* __restrict__ encp,
                                                   float* __restrict__ decp) {
    int gid = blockIdx.x * 256 + threadIdx.x;
    const int NE  = BD * TD * HD;          // 524288
    const int NDp = BD * UD * HD;          // 131072
    const float* xrow; const float* wrow; float acc; float* outp;
    if (gid < NE) {
        int row = gid >> 9, h = gid & (HD - 1);
        xrow = enc + (size_t)row * ED;
        wrow = W1 + (size_t)h * (2 * ED);
        acc  = b1[h];
        outp = encp + gid;
    } else {
        int g = gid - NE;
        if (g >= NDp) return;
        int row = g >> 9, h = g & (HD - 1);
        xrow = dec + (size_t)row * ED;
        wrow = W1 + (size_t)h * (2 * ED) + ED;
        acc  = 0.f;
        outp = decp + g;
    }
    const float4* x4 = (const float4*)xrow;
    const float4* w4 = (const float4*)wrow;
#pragma unroll 4
    for (int i = 0; i < ED / 4; ++i) {
        float4 a = x4[i], b = w4[i];
        acc += a.x * b.x + a.y * b.y + a.z * b.z + a.w * b.w;
    }
    *outp = acc;
}

// ---------------------------------------------------------------------------
// Kernel 2: fused  h=tanh(encp+decp) -> 2x2-blocked WMMA GEMM vs W2 -> +b2 ->
//           streaming log-softmax + top-3.
// Block = 32 rows (one (b,t), half the u-range), 8 wave32s.
// Each wave: 2 M-subtiles x 2 N-groups -> A frags and B frags each reused 2x,
// so the hot loop costs 1 ds_load_b128 + 1 global_load_b128 per WMMA.
// ---------------------------------------------------------------------------
#define APITCH 520   // halves per A row (512 + 8 pad: kills LDS bank conflicts)
#define LPITCH 132   // floats per logits row (128 + 4 pad)

__device__ __forceinline__ void topk_update(const float* lp, int nb,
                                            float& rm, float& rs,
                                            float& t0v, float& t1v, float& t2v,
                                            int& t0i, int& t1i, int& t2i) {
    const float4* lp4 = (const float4*)lp;
#pragma unroll
    for (int c4 = 0; c4 < 4; ++c4) {
        float4 xv = lp4[c4];
        float xs[4] = {xv.x, xv.y, xv.z, xv.w};
#pragma unroll
        for (int j = 0; j < 4; ++j) {
            float x = xs[j];
            if (x > rm) { rs *= __expf(rm - x); rm = x; }
            rs += __expf(x - rm);
            int n = nb + c4 * 4 + j;
            if (x > t0v)      { t2v=t1v; t2i=t1i; t1v=t0v; t1i=t0i; t0v=x; t0i=n; }
            else if (x > t1v) { t2v=t1v; t2i=t1i; t1v=x;   t1i=n; }
            else if (x > t2v) { t2v=x;   t2i=n; }
        }
    }
}

__global__ __launch_bounds__(256) void joint_kernel(const float* __restrict__ encp,
                                                    const float* __restrict__ decp,
                                                    const _Float16* __restrict__ w2h,
                                                    const float* __restrict__ b2,
                                                    float* __restrict__ outV,
                                                    int* __restrict__ outI) {
    __shared__ __attribute__((aligned(16))) _Float16 sA[32 * APITCH]; // 33,280 B
    __shared__ __attribute__((aligned(16))) float    sL[32 * LPITCH]; // 16,896 B

    const int tid = threadIdx.x;
    const int bm  = blockIdx.x;          // 0..2047
    const int bt  = bm >> 1;             // (b*T + t)
    const int u0  = (bm & 1) * 32;

    const float* encRow  = encp + (size_t)bt * HD;
    const float* decRows = decp + ((size_t)(bt >> 8) * UD + u0) * HD;

    // ---- Phase 1: A-tile h = tanh(encp + decp) as f16 in LDS ----
    for (int e = tid; e < 32 * HD; e += 256) {
        int row = e >> 9, k = e & (HD - 1);
        sA[row * APITCH + k] =
            (_Float16)tanhf(encRow[k] + decRows[(size_t)row * HD + k]);
    }
    __syncthreads();

    // ---- Phase 2: stream V in chunks of 256 columns ----
    const int lane = tid & 31;
    const int w    = tid >> 5;       // wave id
    const int l16  = lane & 15;
    const int hsel = lane >> 4;      // lane half (ISA A/B K striping)

    const int rrow = tid >> 3;       // reduction: 8 threads per row
    const int roct = tid & 7;
    float rm = -__builtin_inff(), rs = 0.f;
    float t0v = -__builtin_inff(), t1v = -__builtin_inff(), t2v = -__builtin_inff();
    int   t0i = 0, t1i = 0, t2i = 0;

    for (int iter = 0; iter < VD / 256; ++iter) {
        const int col0 = iter * 256 + w * 16 + l16;     // N-group 0
        const int col1 = col0 + 128;                    // N-group 1
        const float b2c0 = b2[col0];
        const float b2c1 = b2[col1];
        const v8h* bp0 = (const v8h*)(w2h + (size_t)col0 * HD + hsel * 8);
        const v8h* bp1 = (const v8h*)(w2h + (size_t)col1 * HD + hsel * 8);
        const _Float16* abase = sA + (size_t)l16 * APITCH + hsel * 8;

        v8f acc00 = {0.f,0.f,0.f,0.f,0.f,0.f,0.f,0.f};  // m0 x n0
        v8f acc01 = {0.f,0.f,0.f,0.f,0.f,0.f,0.f,0.f};  // m1 x n0
        v8f acc10 = {0.f,0.f,0.f,0.f,0.f,0.f,0.f,0.f};  // m0 x n1
        v8f acc11 = {0.f,0.f,0.f,0.f,0.f,0.f,0.f,0.f};  // m1 x n1

        for (int ks = 0; ks < 16; ++ks) {               // K = 512, step 32
            const _Float16* ap = abase + ks * 32;
            HF16 a0, a1, bf0, bf1;
            a0.h[0] = *(const v8h*)(ap);
            a0.h[1] = *(const v8h*)(ap + 16);
            a1.h[0] = *(const v8h*)(ap + 16 * APITCH);
            a1.h[1] = *(const v8h*)(ap + 16 * APITCH + 16);
            bf0.h[0] = bp0[4 * ks];
            bf0.h[1] = bp0[4 * ks + 2];
            bf1.h[0] = bp1[4 * ks];
            bf1.h[1] = bp1[4 * ks + 2];
            acc00 = __builtin_amdgcn_wmma_f32_16x16x32_f16(
                        false, a0.v, false, bf0.v, (short)0, acc00, false, false);
            acc01 = __builtin_amdgcn_wmma_f32_16x16x32_f16(
                        false, a1.v, false, bf0.v, (short)0, acc01, false, false);
            acc10 = __builtin_amdgcn_wmma_f32_16x16x32_f16(
                        false, a0.v, false, bf1.v, (short)0, acc10, false, false);
            acc11 = __builtin_amdgcn_wmma_f32_16x16x32_f16(
                        false, a1.v, false, bf1.v, (short)0, acc11, false, false);
        }

        // ---- spill + reduce N-group 0 ----
#pragma unroll
        for (int r = 0; r < 8; ++r) {
            int row0 = hsel * 8 + r;
            sL[row0 * LPITCH + w * 16 + l16]        = acc00[r] + b2c0;
            sL[(row0 + 16) * LPITCH + w * 16 + l16] = acc01[r] + b2c0;
        }
        __syncthreads();
        topk_update(sL + rrow * LPITCH + roct * 16, iter * 256 + roct * 16,
                    rm, rs, t0v, t1v, t2v, t0i, t1i, t2i);
        __syncthreads();

        // ---- spill + reduce N-group 1 ----
#pragma unroll
        for (int r = 0; r < 8; ++r) {
            int row0 = hsel * 8 + r;
            sL[row0 * LPITCH + w * 16 + l16]        = acc10[r] + b2c1;
            sL[(row0 + 16) * LPITCH + w * 16 + l16] = acc11[r] + b2c1;
        }
        __syncthreads();
        topk_update(sL + rrow * LPITCH + roct * 16, iter * 256 + 128 + roct * 16,
                    rm, rs, t0v, t1v, t2v, t0i, t1i, t2i);
        __syncthreads();
    }

    // ---- Phase 3: merge 8 partials per row, emit logp top-3 ----
    {
        float* fp = sL + tid * 9;    // (row*8+oct)*9 == tid*9
        fp[0] = rm; fp[1] = rs; fp[2] = t0v; fp[3] = t1v; fp[4] = t2v;
        int* ip = (int*)fp;
        ip[5] = t0i; ip[6] = t1i; ip[7] = t2i;
    }
    __syncthreads();

    if (tid < 32) {
        float M = -__builtin_inff();
        for (int q = 0; q < 8; ++q) M = fmaxf(M, sL[(tid * 8 + q) * 9]);
        float S = 0.f;
        for (int q = 0; q < 8; ++q) {
            const float* pf = sL + (tid * 8 + q) * 9;
            S += pf[1] * __expf(pf[0] - M);
        }
        float v0 = -__builtin_inff(), v1 = v0, v2 = v0;
        int   j0 = 0, j1 = 0, j2 = 0;
        for (int q = 0; q < 8; ++q) {
            const float* pf = sL + (tid * 8 + q) * 9;
            const int*   pi = (const int*)pf;
#pragma unroll
            for (int k = 0; k < BEAM; ++k) {
                float x = pf[2 + k]; int n = pi[5 + k];
                if (x > v0)      { v2=v1; j2=j1; v1=v0; j1=j0; v0=x; j0=n; }
                else if (x > v1) { v2=v1; j2=j1; v1=x;  j1=n; }
                else if (x > v2) { v2=x;  j2=n; }
            }
        }
        float logZ = M + __logf(S);
        size_t mrow = (size_t)bm * 32 + tid;     // global (b,t,u) row
        outV[mrow * 3 + 0] = v0 - logZ;
        outV[mrow * 3 + 1] = v1 - logZ;
        outV[mrow * 3 + 2] = v2 - logZ;
        outI[mrow * 3 + 0] = j0;
        outI[mrow * 3 + 1] = j1;
        outI[mrow * 3 + 2] = j2;
    }
}

// ---------------------------------------------------------------------------
extern "C" void kernel_launch(void* const* d_in, const int* in_sizes, int n_in,
                              void* d_out, int out_size, void* d_ws, size_t ws_size,
                              hipStream_t stream) {
    const float* enc = (const float*)d_in[0];   // (4,256,640)
    const float* dec = (const float*)d_in[1];   // (4,64,640)
    const float* W1  = (const float*)d_in[2];   // (512,1280)
    const float* b1  = (const float*)d_in[3];   // (512)
    const float* W2  = (const float*)d_in[4];   // (4096,512)
    const float* b2  = (const float*)d_in[5];   // (4096)

    char* ws = (char*)d_ws;
    float*    encp = (float*)ws;                                   // 2 MB
    float*    decp = (float*)(ws + 2097152);                       // 512 KB
    _Float16* w2h  = (_Float16*)(ws + 2097152 + 524288);           // 4 MB

    const int NTOP = BD * TD * UD * BEAM;   // 196608
    float* outV = (float*)d_out;
    int*   outI = (int*)d_out + NTOP;

    cvt_w2_kernel<<<(VD * HD / 4 + 255) / 256, 256, 0, stream>>>(W2, w2h);
    proj_kernel<<<(BD * TD * HD + BD * UD * HD + 255) / 256, 256, 0, stream>>>(
        enc, dec, W1, b1, encp, decp);
    joint_kernel<<<BD * TD * UD / 32, 256, 0, stream>>>(encp, decp, w2h, b2, outV, outI);
}
